// BasicLSTM_64055142252654
// MI455X (gfx1250) — compile-verified, matrix operand order checked by reading, not run on hardware
//
#include <hip/hip_runtime.h>

typedef __attribute__((ext_vector_type(16))) _Float16 v16h;
typedef __attribute__((ext_vector_type(8)))  float    v8f;

union AF { v16h v; uint4 q[2]; };

#define VOCAB 32000
#define EDIM  128
#define HDIM  128
#define BATCH 256
#define TLEN  512
// comb width = E + H = 256

// ---- per-lane/per-half K mapping for 16-bit 16x32 WMMA operands -------------
// From CDNA5 ISA "16-bit A-Matrix 16x32":
//   lanes 0-15: M=lane, VGPR0..3 hold K=0..7 (pairs), VGPR4..7 hold K=16..23
//   lanes 16-31: same M, K offset +8  (K=8..15 / K=24..31)
// B (32x16) uses the mirrored layout with lane%16 = N column.
__device__ __forceinline__ int kmap(int h, int lane) {
    int v = h >> 1, sub = h & 1;
    int kl = (v < 4) ? (2 * v + sub) : (16 + 2 * (v - 4) + sub);
    return kl + ((lane & 16) ? 8 : 0);
}

// ---- kernel 1: embedding gather + f32->f16, layout [T][B][E] ----------------
__global__ void __launch_bounds__(256)
gather_embed(const int* __restrict__ x, const float* __restrict__ emb,
             _Float16* __restrict__ xe) {
    int gw   = (blockIdx.x * blockDim.x + threadIdx.x) >> 5;  // row = t*256+b
    int lane = threadIdx.x & 31;
    int t = gw >> 8, b = gw & 255;
    int idx = x[b * TLEN + t];                 // x is [B][T]
    const float4* src = (const float4*)(emb + (size_t)idx * EDIM);
    float4 v = src[lane];                      // 32 lanes * 4 = 128 floats
    union { _Float16 h[4]; uint2 u; } pk;
    pk.h[0] = (_Float16)v.x; pk.h[1] = (_Float16)v.y;
    pk.h[2] = (_Float16)v.z; pk.h[3] = (_Float16)v.w;
    ((uint2*)(xe + (size_t)gw * EDIM))[lane] = pk.u;
}

// ---- kernel 2: pack gate weights + Wout into f16 WMMA-B fragments -----------
// gate frag index f = (g*8 + kt)*8 + nt,  512 halfs each (32 lanes * 16)
// wout frag index f = nt*4 + kt
__global__ void __launch_bounds__(256)
pack_weights(const float* __restrict__ Wi, const float* __restrict__ Wf,
             const float* __restrict__ Wc, const float* __restrict__ Wo,
             const float* __restrict__ Wout,
             _Float16* __restrict__ wg, _Float16* __restrict__ woutF) {
    int gw   = (blockIdx.x * blockDim.x + threadIdx.x) >> 5;
    int lane = threadIdx.x & 31;
    AF frag;
    if (gw < 256) {                                  // 4 gates * 8 kt * 8 nt
        int g = gw >> 6, kt = (gw >> 3) & 7, nt = gw & 7;
        const float* W = (g == 0) ? Wi : (g == 1) ? Wf : (g == 2) ? Wc : Wo;
        int n = nt * 16 + (lane & 15);
#pragma unroll
        for (int h = 0; h < 16; ++h) {
            int k = kt * 32 + kmap(h, lane);
            frag.v[h] = (_Float16)W[k * HDIM + n];   // W is [256][128]
        }
        uint4* dst = (uint4*)(wg + (size_t)gw * 512);
        dst[lane * 2] = frag.q[0]; dst[lane * 2 + 1] = frag.q[1];
    } else if (gw < 256 + 8000) {                    // 2000 nt * 4 kt
        int f = gw - 256;
        int nt = f >> 2, kt = f & 3;
        int n = nt * 16 + (lane & 15);
#pragma unroll
        for (int h = 0; h < 16; ++h) {
            int k = kt * 32 + kmap(h, lane);
            frag.v[h] = (_Float16)Wout[(size_t)k * VOCAB + n];  // [128][32000]
        }
        uint4* dst = (uint4*)(woutF + (size_t)f * 512);
        dst[lane * 2] = frag.q[0]; dst[lane * 2 + 1] = frag.q[1];
    }
}

// ---- kernel 3: persistent LSTM scan, 16 WGs x 16 batch rows, 512 thr --------
// Weights live in REGISTERS (128 VGPRs/wave of B-fragments, loop-invariant).
// dynamic LDS layout (bytes):
//   [0,      8192) comb [16][256] f16   (cols 0-127 = e_t, 128-255 = h)
//   [8192,  40960) gate pre-acts [4][16][128] f32
//   [40960, 49152) cell state c [16][128] f32
//   [49152, 51200) biases [4][128] f32
#define LSTM_LDS 51200
__global__ void __launch_bounds__(512, 1)
lstm_scan(const _Float16* __restrict__ xe, const _Float16* __restrict__ wg,
          const float* __restrict__ bi, const float* __restrict__ bf,
          const float* __restrict__ bc, const float* __restrict__ bo,
          float* __restrict__ hfinal) {
    extern __shared__ __align__(16) char smem[];
    _Float16* sComb = (_Float16*)smem;
    float*    sGate = (float*)(smem + 8192);
    float*    sC    = (float*)(smem + 40960);
    float*    sBias = (float*)(smem + 49152);

    const int tid  = threadIdx.x;
    const int lane = tid & 31;
    const int w    = tid >> 5;                 // wave 0..15
    const int b0   = blockIdx.x * 16;          // batch slice base

    const int g      = w & 3;                  // gate this wave computes
    const int ntBase = (w >> 2) * 2;           // 2 n-tiles per wave
    const int m      = lane & 15;
    const int koffQ  = (lane & 16) ? 1 : 0;    // +8 halfs = +1 uint4
    const int mo     = (lane & 16) ? 8 : 0;

    // loop-invariant weight B-fragments -> registers (32 x b128 loads, L2 hit)
    AF breg[2][8];
    {
        const uint4* wq = (const uint4*)wg;
#pragma unroll
        for (int jj = 0; jj < 2; ++jj)
#pragma unroll
            for (int kt = 0; kt < 8; ++kt) {
                int f = (g * 8 + kt) * 8 + (ntBase + jj);
                breg[jj][kt].q[0] = wq[f * 64 + lane * 2];
                breg[jj][kt].q[1] = wq[f * 64 + lane * 2 + 1];
            }
    }

    sBias[tid] = (tid < 128) ? bi[tid]
               : (tid < 256) ? bf[tid - 128]
               : (tid < 384) ? bc[tid - 256] : bo[tid - 384];
#pragma unroll
    for (int i = tid; i < 2048; i += 512) {
        sC[i] = 0.0f;
        sComb[(i >> 7) * 256 + 128 + (i & 127)] = (_Float16)0.0f;  // h0 = 0
    }
    __syncthreads();

    // e_t pipeline: threads 0..255 each own one uint4 of the 4KB slice
    const uint4* xeq = (const uint4*)xe + ((size_t)b0 + (tid >> 4)) * 16 + (tid & 15);
    const int    cq  = (tid >> 4) * 32 + (tid & 15);  // comb uint4 index
    uint4 ecur;
    if (tid < 256) ecur = xeq[0];

    for (int t = 0; t < TLEN; ++t) {
        // phase 0: stage e_t into comb[:, :128]
        if (tid < 256) ((uint4*)sComb)[cq] = ecur;
        __syncthreads();

        // prefetch e_{t+1} (latency overlaps GEMM + elementwise)
        if (tid < 256 && t + 1 < TLEN) ecur = xeq[(size_t)(t + 1) * 4096];

        // phase 1: gate GEMMs, M=16 N=32 K=256 per wave, weights in registers
        const uint4* combQ = (const uint4*)sComb;
        v8f acc0 = {}, acc1 = {};
#pragma unroll
        for (int kt = 0; kt < 8; ++kt) {
            AF a;
            int bq = m * 32 + kt * 4 + koffQ;
            a.q[0] = combQ[bq];          // K 0..7   (+koff)
            a.q[1] = combQ[bq + 2];      // K 16..23 (+koff)
            acc0 = __builtin_amdgcn_wmma_f32_16x16x32_f16(
                false, a.v, false, breg[0][kt].v, (short)0, acc0, false, false);
            acc1 = __builtin_amdgcn_wmma_f32_16x16x32_f16(
                false, a.v, false, breg[1][kt].v, (short)0, acc1, false, false);
        }
        {
            float* gp = sGate + g * 2048;
            int n0 = ntBase * 16 + (lane & 15);
#pragma unroll
            for (int r = 0; r < 8; ++r) gp[(r + mo) * 128 + n0] = acc0[r];
#pragma unroll
            for (int r = 0; r < 8; ++r) gp[(r + mo) * 128 + n0 + 16] = acc1[r];
        }
        __syncthreads();

        // phase 2: elementwise gate math, 4 elements per thread
        const bool last = (t == TLEN - 1);
#pragma unroll
        for (int q = 0; q < 4; ++q) {
            int e  = tid * 4 + q;
            int mm = e >> 7, nn = e & 127;
            float gi = sGate[e]          + sBias[nn];
            float gf = sGate[2048 + e]   + sBias[128 + nn];
            float gc = sGate[4096 + e]   + sBias[256 + nn];
            float go = sGate[6144 + e]   + sBias[384 + nn];
            float si = 1.0f / (1.0f + __expf(-gi));
            float sf = 1.0f / (1.0f + __expf(-gf));
            float so = 1.0f / (1.0f + __expf(-go));
            float cb = 2.0f / (1.0f + __expf(-2.0f * gc)) - 1.0f;   // tanh
            float c  = sf * sC[e] + si * cb;
            sC[e] = c;
            float th = 2.0f / (1.0f + __expf(-2.0f * c)) - 1.0f;    // tanh
            float h  = so * th;
            sComb[mm * 256 + 128 + nn] = (_Float16)h;               // h for t+1
            if (last) hfinal[(size_t)(b0 + mm) * HDIM + nn] = h;
        }
        __syncthreads();
    }
}

// ---- kernel 4: logits = hfinal @ Wout + bout, WMMA f16 ----------------------
__global__ void __launch_bounds__(256)
proj_logits(const float* __restrict__ hfinal, const _Float16* __restrict__ woutF,
            const float* __restrict__ bout, float* __restrict__ out) {
    int lane = threadIdx.x & 31;
    int w    = threadIdx.x >> 5;
    int mt   = blockIdx.x & 15;            // 16 M tiles
    int nt   = (blockIdx.x >> 4) * 8 + w;  // 2000 N tiles
    int mrow = mt * 16 + (lane & 15);
    AF a[4];
#pragma unroll
    for (int kt = 0; kt < 4; ++kt)
#pragma unroll
        for (int h = 0; h < 16; ++h)
            a[kt].v[h] = (_Float16)hfinal[(size_t)mrow * HDIM + kt * 32 + kmap(h, lane)];
    const uint4* wq = (const uint4*)woutF;
    v8f acc = {};
#pragma unroll
    for (int kt = 0; kt < 4; ++kt) {
        AF bfr;
        size_t f = (size_t)nt * 4 + kt;
        bfr.q[0] = wq[f * 64 + lane * 2];
        bfr.q[1] = wq[f * 64 + lane * 2 + 1];
        acc = __builtin_amdgcn_wmma_f32_16x16x32_f16(
            false, a[kt].v, false, bfr.v, (short)0, acc, false, false);
    }
    int   n  = nt * 16 + (lane & 15);
    float bb = bout[n];
    int   mo = (lane & 16) ? 8 : 0;
#pragma unroll
    for (int r = 0; r < 8; ++r)
        out[(size_t)(mt * 16 + r + mo) * VOCAB + n] = acc[r] + bb;
}

extern "C" void kernel_launch(void* const* d_in, const int* in_sizes, int n_in,
                              void* d_out, int out_size, void* d_ws, size_t ws_size,
                              hipStream_t stream) {
    (void)in_sizes; (void)n_in; (void)out_size; (void)ws_size;
    const int*   x    = (const int*)d_in[0];
    const float* emb  = (const float*)d_in[1];
    const float* Wi   = (const float*)d_in[2];
    const float* bi   = (const float*)d_in[3];
    const float* Wf   = (const float*)d_in[4];
    const float* bf   = (const float*)d_in[5];
    const float* Wc   = (const float*)d_in[6];
    const float* bc   = (const float*)d_in[7];
    const float* Wo   = (const float*)d_in[8];
    const float* bo   = (const float*)d_in[9];
    const float* Wout = (const float*)d_in[10];
    const float* bout = (const float*)d_in[11];
    float*       out  = (float*)d_out;

    char* ws = (char*)d_ws;
    _Float16* xe     = (_Float16*)ws;                   // 33,554,432 B
    _Float16* wg     = (_Float16*)(ws + 33554432);      //    262,144 B
    _Float16* woutF  = (_Float16*)(ws + 33816576);      //  8,192,000 B
    float*    hfinal = (float*)   (ws + 42008576);      //    131,072 B

    gather_embed<<<16384, 256, 0, stream>>>(x, emb, xe);          // 131072 rows
    pack_weights<<<1032, 256, 0, stream>>>(Wi, Wf, Wc, Wo, Wout, wg, woutF);
    hipError_t attr_rc = hipFuncSetAttribute(
        (const void*)lstm_scan, hipFuncAttributeMaxDynamicSharedMemorySize, LSTM_LDS);
    (void)attr_rc;
    lstm_scan<<<16, 512, LSTM_LDS, stream>>>(xe, wg, bi, bf, bc, bo, hfinal);
    proj_logits<<<4000, 256, 0, stream>>>(hfinal, woutF, bout, out);
}
